// GuidedCxtAttenEmbedding_55688545960151
// MI455X (gfx1250) — matrix operand order repaired and checked by reading, not compiled, not run on hardware
//
#include <hip/hip_runtime.h>
#include <hip/hip_bf16.h>
#include <math.h>

typedef __bf16 bf16_t;
typedef __attribute__((ext_vector_type(16))) __bf16 v16bf;
typedef __attribute__((ext_vector_type(8)))  __bf16 v8bf;
typedef __attribute__((ext_vector_type(8)))  float  v8f;
typedef unsigned int v4u __attribute__((ext_vector_type(4)));
typedef unsigned int v8u __attribute__((ext_vector_type(8)));

#define BB   4      // batch
#define GCH  128    // input feature channels
#define CAH  32     // alpha channels
#define HF   128    // full-res H/W
#define HH   64     // half-res H/W
#define NN   4096   // HH*HH
#define CC   64     // fg channels
#define KD   576    // 64*3*3 patch dim (QK inner dim)
#define VD   512    // 32*4*4 value dim

__device__ __forceinline__ int iabs_(int x) { return x < 0 ? -x : x; }
__device__ __forceinline__ int imin_(int a, int b) { return a < b ? a : b; }
__device__ __forceinline__ int imax_(int a, int b) { return a > b ? a : b; }
__device__ __forceinline__ int reflect64(int i)  { return (i < 0) ? -i : ((i >= HH) ? (2*HH - 2 - i) : i); }
__device__ __forceinline__ int reflect128(int i) { return (i < 0) ? -i : ((i >= HF) ? (2*HF - 2 - i) : i); }

// ---------------------------------------------------------------------------
// Tensor Data Mover: issue a 2D tile DMA (global -> LDS), bf16 elements.
// D# group0/group1 bitfields per CDNA5 ISA §8.3/§8.4; 2-group form (VADDR2/3
// NULL => tensors up to 2D).  Tracked by TENSORcnt.
// ---------------------------------------------------------------------------
__device__ __forceinline__ void tdm_load_2d(unsigned lds_off, const void* gaddr,
                                            unsigned tile_d0, unsigned tile_d1,
                                            unsigned tensor_d0, unsigned tensor_d1,
                                            unsigned stride_d0)
{
    const unsigned long long ga = (unsigned long long)(uintptr_t)gaddr;
    v4u g0;
    g0[0] = 1u;                                                   // count=1 (valid user D#)
    g0[1] = lds_off;                                              // lds_addr (bytes)
    g0[2] = (unsigned)ga;                                         // global_addr[31:0]
    g0[3] = ((unsigned)(ga >> 32) & 0x01FFFFFFu) | (2u << 30);    // global_addr[56:32] | type=2
    v8u g1;
    g1[0] = 1u << 16;                                             // wg_mask=0, data_size=1 (2B)
    g1[1] = (tensor_d0 & 0xFFFFu) << 16;                          // tensor_dim0[15:0]
    g1[2] = (tensor_d0 >> 16) | ((tensor_d1 & 0xFFFFu) << 16);    // dim0[31:16] | dim1[15:0]
    g1[3] = (tensor_d1 >> 16) | (tile_d0 << 16);                  // dim1[31:16] | tile_dim0
    g1[4] = tile_d1 & 0xFFFFu;                                    // tile_dim1 (tile_dim2=0)
    g1[5] = stride_d0;                                            // tensor_dim0_stride[31:0]
    g1[6] = 0u;                                                   // stride hi / dim1_stride lo
    g1[7] = 0u;
    asm volatile("tensor_load_to_lds %0, %1" :: "s"(g0), "s"(g1) : "memory");
}

// ---------------------------------------------------------------------------
// K0: one block. E = pos_emb @ lw^T + lb ; spectral-norm+BN folded coefs ; um=0
// ---------------------------------------------------------------------------
__global__ void prep_const_kernel(const float* __restrict__ lw, const float* __restrict__ lb,
                                  const float* __restrict__ w_bar,
                                  const float* __restrict__ uvec, const float* __restrict__ vvec,
                                  const float* __restrict__ bn_g, const float* __restrict__ bn_b,
                                  const float* __restrict__ bn_m, const float* __restrict__ bn_v,
                                  float* __restrict__ E, float* __restrict__ coefA,
                                  float* __restrict__ coefB, float* __restrict__ um)
{
    __shared__ float sig_sh;
    const int t = threadIdx.x;
    if (t < BB) um[t] = 0.f;
    if (t == 0) {
        float s = 0.f;
        for (int o = 0; o < CAH; ++o) {
            float r = 0.f;
            for (int c = 0; c < CAH; ++c) r += w_bar[o*CAH + c] * vvec[c];
            s += uvec[o] * r;
        }
        sig_sh = s;
    }
    __syncthreads();
    // E[d][c], d in [0,8)
    for (int e = t; e < 8*CC; e += 256) {
        const int d = e / CC, c = e % CC;
        const float pos = (float)(7 - d);
        float acc = lb[c];
        for (int j = 0; j < CC; ++j) {
            const int jj = (j < 32) ? j : (j - 32);
            const float invf = __expf(-((float)(2*jj) / (float)CC) * 9.2103403719761836f); // ln(1e4)
            const float sarg = pos * invf;
            const float pe = (j < 32) ? __sinf(sarg) : __cosf(sarg);
            acc += pe * lw[c*CC + j];
        }
        E[d*CC + c] = acc;
    }
    const float sig = sig_sh;
    for (int e = t; e < CAH*CAH; e += 256) {
        const int o = e / CAH, c = e % CAH;
        const float g = bn_g[o] * rsqrtf(bn_v[o] + 1e-5f);
        coefA[e] = (w_bar[e] / sig) * g;
        if (c == 0) coefB[o] = bn_b[o] - bn_m[o] * g;
    }
}

// ---------------------------------------------------------------------------
// K1: fg[b,:,h,w] = gw @ f[b,:,2h,2w] + gb (channel-last), unknown + um accum
// ---------------------------------------------------------------------------
__global__ void fg_kernel(const float* __restrict__ f, const float* __restrict__ trimap,
                          const float* __restrict__ gw, const float* __restrict__ gb,
                          float* __restrict__ fg_cl, float* __restrict__ unk, float* __restrict__ um)
{
    __shared__ float fsh[GCH];
    const int id = blockIdx.x;            // b*NN + n
    const int b = id >> 12, n = id & 4095;
    const int h = n >> 6, w = n & 63;
    const int t = threadIdx.x;            // 64 threads
    const float* fp = f + (size_t)b * GCH * HF * HF + (size_t)(2*h) * HF + (2*w);
    fsh[t]      = fp[(size_t)t * HF * HF];
    fsh[t + 64] = fp[(size_t)(t + 64) * HF * HF];
    __syncthreads();
    float acc = gb[t];
    const float* gr = gw + t * GCH;
    #pragma unroll 8
    for (int i = 0; i < GCH; ++i) acc += gr[i] * fsh[i];
    fg_cl[(size_t)id * CC + t] = acc;
    if (t == 0) {
        const float uv = trimap[((size_t)b*3 + 1) * HF * HF + (size_t)(2*h) * HF + (2*w)];
        unk[id] = uv;
        atomicAdd(&um[b], uv * (1.0f / (float)NN));
    }
}

// ---------------------------------------------------------------------------
// K2: per-batch u_scale/k_scale
// ---------------------------------------------------------------------------
__global__ void scales_kernel(const float* __restrict__ um, float* __restrict__ scales)
{
    const int b = threadIdx.x;
    if (b >= BB) return;
    const float u_ = um[b], k_ = 1.f - u_;
    float us = sqrtf(u_ / k_); us = fminf(fmaxf(us, 0.1f), 10.f);
    float ks = sqrtf(k_ / u_); ks = fminf(fmaxf(ks, 0.1f), 10.f);
    scales[b*2] = us; scales[b*2 + 1] = ks;
}

// ---------------------------------------------------------------------------
// K3: Pq[b,q,0..8) = E @ fg[b,:,q]
// ---------------------------------------------------------------------------
__global__ void pos_kernel(const float* __restrict__ fg_cl, const float* __restrict__ E,
                           float* __restrict__ Pq)
{
    __shared__ float Esh[8*CC];
    const int t = threadIdx.x;
    for (int e = t; e < 8*CC; e += 256) Esh[e] = E[e];
    __syncthreads();
    const int id = blockIdx.x * 256 + t;   // b*NN + n
    const float* src = fg_cl + (size_t)id * CC;
    float acc[8];
    #pragma unroll
    for (int d = 0; d < 8; ++d) acc[d] = 0.f;
    for (int c = 0; c < CC; ++c) {
        const float fv = src[c];
        #pragma unroll
        for (int d = 0; d < 8; ++d) acc[d] += fv * Esh[d*CC + c];
    }
    #pragma unroll
    for (int d = 0; d < 8; ++d) Pq[(size_t)id * 8 + d] = acc[d];
}

// ---------------------------------------------------------------------------
// K4: pack 3x3 reflect patches (Q == K) to bf16, per-key scale & diag mask
// ---------------------------------------------------------------------------
__global__ void qkpack_kernel(const float* __restrict__ fg_cl, const float* __restrict__ unk,
                              const float* __restrict__ scales,
                              bf16_t* __restrict__ Qm, float* __restrict__ ksArr,
                              float* __restrict__ diagArr)
{
    const int id = blockIdx.x * 256 + threadIdx.x;  // b*NN + n
    const int b = id >> 12, n = id & 4095;
    const int h = n >> 6, w = n & 63;
    bf16_t* dst = Qm + (size_t)id * KD;
    float ss = 0.f;
    int pos = 0;
    for (int ky = 0; ky < 3; ++ky) {
        const int yy = reflect64(h + ky - 1);
        for (int kx = 0; kx < 3; ++kx) {
            const int xx = reflect64(w + kx - 1);
            const float* src = fg_cl + ((size_t)b * NN + yy * HH + xx) * CC;
            for (int c = 0; c < CC; ++c) {
                const float fv = src[c];
                ss += fv * fv;
                dst[pos++] = (bf16_t)fv;
            }
        }
    }
    const float inv = 1.f / fmaxf(sqrtf(ss), 1e-4f);
    float m = 0.f;
    for (int ky = 0; ky < 3; ++ky)
        for (int kx = 0; kx < 3; ++kx)
            m += unk[b * NN + reflect64(h + ky - 1) * HH + reflect64(w + kx - 1)];
    m *= (1.f / 9.f);
    const float mm = (m > 0.f) ? 1.f : 0.f;
    const float us = scales[b*2], kss = scales[b*2 + 1];
    ksArr[id]   = inv * (us * mm + kss * (1.f - mm));
    diagArr[id] = -10000.f * mm;
}

// ---------------------------------------------------------------------------
// K5: pack V = aw transposed: awT[b][vcol=o*16+ay*4+ax][n] (bf16)
// ---------------------------------------------------------------------------
__global__ void awpack_kernel(const float* __restrict__ alpha, bf16_t* __restrict__ awT)
{
    const size_t id = (size_t)blockIdx.x * 256 + threadIdx.x;  // b*VD*NN + vcol*NN + n
    const int n    = (int)(id & 4095);
    const int vcol = (int)((id >> 12) & 511);
    const int b    = (int)(id >> 21);
    const int h = n >> 6, w = n & 63;
    const int o = vcol >> 4, ay = (vcol >> 2) & 3, ax = vcol & 3;
    const int Y = reflect128(2*h + ay - 1);
    const int X = reflect128(2*w + ax - 1);
    awT[((size_t)(b * VD + vcol)) * NN + n] =
        (bf16_t)alpha[(((size_t)b * CAH + o) * HF + Y) * HF + X];
}

// ---------------------------------------------------------------------------
// K6: fused flash attention.  One wave = 16 queries x full 512 V-dim (256 f32
// accum VGPRs via CDNA5 VGPR-MSBs).  8 waves / block, block-synchronous key
// loop: K (32x576) and V (512x32) bf16 tiles are DMA'd into LDS by the Tensor
// Data Mover (double-buffered, TENSORcnt-tracked), waves consume via
// ds_load_b128 and feed v_wmma_f32_16x16x32_bf16.
// ---------------------------------------------------------------------------
__global__ __launch_bounds__(256, 1)
void flash_attn_kernel(const bf16_t* __restrict__ Qm, const bf16_t* __restrict__ awT,
                       const float* __restrict__ Pq, const float* __restrict__ ksArr,
                       const float* __restrict__ diagArr, float* __restrict__ Z)
{
    __shared__ float pq_lds[8][16*8];
    __shared__ __align__(16) bf16_t att_lds[8][16*32];
    __shared__ __align__(16) bf16_t ktile[2][32 * KD];   // 2 x 36 KB
    __shared__ __align__(16) bf16_t vtile[2][VD * 32];   // 2 x 32 KB

    const int wave = threadIdx.x >> 5;
    const int lane = threadIdx.x & 31;
    const int col  = lane & 15;     // D/B column, A row
    const int half = lane >> 4;     // half-wave select
    const int row0 = half * 8;      // D rows held by this lane
    const int b    = blockIdx.y;
    const int qbase = (blockIdx.x * 8 + wave) * 16;

    {   // stage this wave's 16 queries x 8 positional biases into LDS
        const float* src = Pq + ((size_t)b * NN + qbase) * 8;
        #pragma unroll
        for (int t = 0; t < 4; ++t) pq_lds[wave][lane*4 + t] = src[lane*4 + t];
    }
    asm volatile("s_wait_dscnt 0" ::: "memory");

    float mrow[8], lrow[8];
    v8f oacc[32];
    #pragma unroll
    for (int v = 0; v < 8; ++v) { mrow[v] = -1e30f; lrow[v] = 0.f; }
    #pragma unroll
    for (int j = 0; j < 32; ++j) {
        #pragma unroll
        for (int v = 0; v < 8; ++v) oacc[j][v] = 0.f;
    }

    const bf16_t* qrow = Qm + ((size_t)b * NN + qbase + col) * KD;  // A row for this lane

    // prime TDM double buffer with key-tile 0
    if (wave == 0) {
        tdm_load_2d((unsigned)(uintptr_t)&ktile[0][0], Qm + ((size_t)b * NN + 0) * KD,
                    KD, 32, KD, NN, KD);
        tdm_load_2d((unsigned)(uintptr_t)&vtile[0][0], awT + (size_t)b * VD * NN + 0,
                    32, VD, NN, VD, NN);
    }

    for (int kt = 0; kt < NN; kt += 32) {
        const int  cur  = (kt >> 5) & 1;
        const bool more = (kt + 32) < NN;
        if (wave == 0) {
            if (more) {   // prefetch next K/V tile pair into the other buffer
                tdm_load_2d((unsigned)(uintptr_t)&ktile[cur ^ 1][0],
                            Qm + ((size_t)b * NN + kt + 32) * KD, KD, 32, KD, NN, KD);
                tdm_load_2d((unsigned)(uintptr_t)&vtile[cur ^ 1][0],
                            awT + (size_t)b * VD * NN + kt + 32, 32, VD, NN, VD, NN);
                asm volatile("s_wait_tensorcnt 0x2" ::: "memory");  // current pair landed
            } else {
                asm volatile("s_wait_tensorcnt 0x0" ::: "memory");
            }
        }
        __syncthreads();

        // ---- S tiles: Q @ K^T over KD, K-fragments from LDS ----
        const bf16_t* krow0 = &ktile[cur][(size_t)col * KD];
        const bf16_t* krow1 = &ktile[cur][(size_t)(col + 16) * KD];
        v8f s0 = {0.f,0.f,0.f,0.f,0.f,0.f,0.f,0.f};
        v8f s1 = {0.f,0.f,0.f,0.f,0.f,0.f,0.f,0.f};
        #pragma unroll 2
        for (int kk = 0; kk < KD; kk += 32) {
            v8bf alo = *(const v8bf*)(qrow + kk + 16*half);
            v8bf ahi = *(const v8bf*)(qrow + kk + 16*half + 8);
            v8bf b0l = *(const v8bf*)(krow0 + kk + 16*half);
            v8bf b0h = *(const v8bf*)(krow0 + kk + 16*half + 8);
            v8bf b1l = *(const v8bf*)(krow1 + kk + 16*half);
            v8bf b1h = *(const v8bf*)(krow1 + kk + 16*half + 8);
            v16bf af, bf0, bf1;
            #pragma unroll
            for (int t = 0; t < 8; ++t) {
                af[t]  = alo[t]; af[t+8]  = ahi[t];
                bf0[t] = b0l[t]; bf0[t+8] = b0h[t];
                bf1[t] = b1l[t]; bf1[t+8] = b1h[t];
            }
            s0 = __builtin_amdgcn_wmma_f32_16x16x32_bf16(false, af, false, bf0, (short)0, s0, false, false);
            s1 = __builtin_amdgcn_wmma_f32_16x16x32_bf16(false, af, false, bf1, (short)0, s1, false, false);
        }

        // ---- per-key scale, diag mask, positional bias ----
        const int key0 = kt + col, key1 = key0 + 16;
        const float ksc0 = ksArr[(size_t)b*NN + key0];
        const float ksc1 = ksArr[(size_t)b*NN + key1];
        const float dg0  = diagArr[(size_t)b*NN + key0];
        const float dg1  = diagArr[(size_t)b*NN + key1];
        const int k0h = key0 >> 6, k0w = key0 & 63;
        const int k1h = key1 >> 6, k1w = key1 & 63;

        float p0[8], p1[8], rmax[8];
        #pragma unroll
        for (int v = 0; v < 8; ++v) {
            const int M = row0 + v;
            const int q = qbase + M;
            const int qh = q >> 6, qw = q & 63;
            int d0 = imin_(imax_(iabs_(qh - k0h), iabs_(qw - k0w)), 7);
            int d1 = imin_(imax_(iabs_(qh - k1h), iabs_(qw - k1w)), 7);
            const float e0 = s0[v]*ksc0 + ((key0 == q) ? dg0 : 0.f) + pq_lds[wave][M*8 + (7 - d0)];
            const float e1 = s1[v]*ksc1 + ((key1 == q) ? dg1 : 0.f) + pq_lds[wave][M*8 + (7 - d1)];
            p0[v] = e0; p1[v] = e1; rmax[v] = fmaxf(e0, e1);
        }
        // row max across the 16 lanes of this half-wave
        #pragma unroll
        for (int v = 0; v < 8; ++v) {
            float r = rmax[v];
            r = fmaxf(r, __shfl_xor(r, 1));
            r = fmaxf(r, __shfl_xor(r, 2));
            r = fmaxf(r, __shfl_xor(r, 4));
            r = fmaxf(r, __shfl_xor(r, 8));
            rmax[v] = r;
        }
        float cf[8];
        #pragma unroll
        for (int v = 0; v < 8; ++v) {
            const float mo = mrow[v];
            const float mn = fmaxf(mo, rmax[v]);
            cf[v] = __expf(mo - mn);
            mrow[v] = mn;
            const float e0 = __expf(p0[v] - mn);
            const float e1 = __expf(p1[v] - mn);
            p0[v] = e0; p1[v] = e1;
            float rs = e0 + e1;
            rs += __shfl_xor(rs, 1);
            rs += __shfl_xor(rs, 2);
            rs += __shfl_xor(rs, 4);
            rs += __shfl_xor(rs, 8);
            lrow[v] = lrow[v] * cf[v] + rs;
        }
        #pragma unroll
        for (int j = 0; j < 32; ++j) {
            #pragma unroll
            for (int v = 0; v < 8; ++v) oacc[j][v] *= cf[v];
        }

        // ---- D-layout att -> LDS -> A-fragment layout (per-wave private) ----
        #pragma unroll
        for (int v = 0; v < 8; ++v) {
            att_lds[wave][(row0 + v)*32 + col]      = (bf16_t)p0[v];
            att_lds[wave][(row0 + v)*32 + col + 16] = (bf16_t)p1[v];
        }
        asm volatile("s_wait_dscnt 0" ::: "memory");
        v16bf attf;
        {
            v8bf t0 = *(const v8bf*)&att_lds[wave][col*32 + 16*half];
            v8bf t1 = *(const v8bf*)&att_lds[wave][col*32 + 16*half + 8];
            #pragma unroll
            for (int t = 0; t < 8; ++t) { attf[t] = t0[t]; attf[t+8] = t1[t]; }
        }

        // ---- O += att @ V, V-fragments from the TDM-staged LDS tile ----
        #pragma unroll
        for (int j = 0; j < 32; ++j) {
            const bf16_t* vp = &vtile[cur][(size_t)(j*16 + col) * 32 + 16*half];
            v8bf v0 = *(const v8bf*)vp;
            v8bf v1 = *(const v8bf*)(vp + 8);
            v16bf bfv;
            #pragma unroll
            for (int t = 0; t < 8; ++t) { bfv[t] = v0[t]; bfv[t+8] = v1[t]; }
            oacc[j] = __builtin_amdgcn_wmma_f32_16x16x32_bf16(false, attf, false, bfv, (short)0, oacc[j], false, false);
        }
        __syncthreads();   // all waves done with buffer `cur` before it is refilled
    }

    // ---- epilogue: Z = O / l ----
    #pragma unroll
    for (int j = 0; j < 32; ++j) {
        #pragma unroll
        for (int v = 0; v < 8; ++v) {
            const int q = qbase + row0 + v;
            Z[((size_t)b * NN + q) * VD + j*16 + col] = oacc[j][v] / lrow[v];
        }
    }
}

// ---------------------------------------------------------------------------
// K7: overlap-add paste (/4), spectral-norm 1x1 conv + BN, + alpha
// ---------------------------------------------------------------------------
__global__ void paste_kernel(const float* __restrict__ Z, const float* __restrict__ coefA,
                             const float* __restrict__ coefB, const float* __restrict__ alpha,
                             float* __restrict__ out)
{
    __shared__ float cA[CAH*CAH];
    __shared__ float cB[CAH];
    const int t = threadIdx.x;
    for (int e = t; e < CAH*CAH; e += 256) cA[e] = coefA[e];
    if (t < CAH) cB[t] = coefB[t];
    __syncthreads();

    const int id = blockIdx.x * 256 + t;   // b*HF*HF + Y*HF + X
    const int b = id >> 14;
    const int px = id & 16383;
    const int Y = px >> 7, X = px & 127;

    float ypre[CAH];
    #pragma unroll
    for (int c = 0; c < CAH; ++c) ypre[c] = 0.f;

    const int py = (Y + 1) & 1, pxp = (X + 1) & 1;
    for (int ai = 0; ai < 2; ++ai) {
        const int ay = py + 2*ai;
        const int hh2 = Y + 1 - ay;
        if (hh2 < 0) continue;
        const int h = hh2 >> 1;
        if (h >= HH) continue;
        for (int aj = 0; aj < 2; ++aj) {
            const int ax = pxp + 2*aj;
            const int ww2 = X + 1 - ax;
            if (ww2 < 0) continue;
            const int w = ww2 >> 1;
            if (w >= HH) continue;
            const float* zp = Z + ((size_t)b * NN + h * HH + w) * VD + (ay*4 + ax);
            #pragma unroll
            for (int c = 0; c < CAH; ++c) ypre[c] += zp[c * 16];
        }
    }
    #pragma unroll
    for (int c = 0; c < CAH; ++c) ypre[c] *= 0.25f;

    for (int o = 0; o < CAH; ++o) {
        float acc = cB[o];
        #pragma unroll
        for (int c = 0; c < CAH; ++c) acc += cA[o*CAH + c] * ypre[c];
        const size_t oi = (((size_t)b * CAH + o) * HF + Y) * HF + X;
        out[oi] = acc + alpha[oi];
    }
}

// ---------------------------------------------------------------------------
extern "C" void kernel_launch(void* const* d_in, const int* in_sizes, int n_in,
                              void* d_out, int out_size, void* d_ws, size_t ws_size,
                              hipStream_t stream)
{
    (void)in_sizes; (void)n_in; (void)out_size; (void)ws_size;
    const float* f      = (const float*)d_in[0];
    const float* alpha  = (const float*)d_in[1];
    const float* trimap = (const float*)d_in[2];
    const float* gw     = (const float*)d_in[3];
    const float* gb     = (const float*)d_in[4];
    const float* lw     = (const float*)d_in[5];
    const float* lb     = (const float*)d_in[6];
    const float* w_bar  = (const float*)d_in[7];
    const float* uvec   = (const float*)d_in[8];
    const float* vvec   = (const float*)d_in[9];
    const float* bn_g   = (const float*)d_in[10];
    const float* bn_b   = (const float*)d_in[11];
    const float* bn_m   = (const float*)d_in[12];
    const float* bn_v   = (const float*)d_in[13];
    float* out = (float*)d_out;

    uint8_t* base = (uint8_t*)d_ws;
    size_t off = 0;
    auto carve = [&](size_t bytes) -> void* {
        void* p = base + off;
        off = (off + bytes + 255) & ~(size_t)255;
        return p;
    };
    // total ~74 MB
    float*  fg_cl  = (float*) carve((size_t)BB*NN*CC*4);
    float*  unk    = (float*) carve((size_t)BB*NN*4);
    float*  um     = (float*) carve((size_t)BB*4);
    float*  scales = (float*) carve((size_t)BB*2*4);
    float*  E      = (float*) carve((size_t)8*CC*4);
    float*  coefA  = (float*) carve((size_t)CAH*CAH*4);
    float*  coefB  = (float*) carve((size_t)CAH*4);
    float*  Pq     = (float*) carve((size_t)BB*NN*8*4);
    float*  ksArr  = (float*) carve((size_t)BB*NN*4);
    float*  diagA  = (float*) carve((size_t)BB*NN*4);
    bf16_t* Qm     = (bf16_t*)carve((size_t)BB*NN*KD*2);
    bf16_t* awT    = (bf16_t*)carve((size_t)BB*VD*NN*2);
    float*  Z      = (float*) carve((size_t)BB*NN*VD*4);

    prep_const_kernel<<<1, 256, 0, stream>>>(lw, lb, w_bar, uvec, vvec,
                                             bn_g, bn_b, bn_m, bn_v, E, coefA, coefB, um);
    fg_kernel<<<BB*NN, 64, 0, stream>>>(f, trimap, gw, gb, fg_cl, unk, um);
    scales_kernel<<<1, 32, 0, stream>>>(um, scales);
    pos_kernel<<<(BB*NN)/256, 256, 0, stream>>>(fg_cl, E, Pq);
    qkpack_kernel<<<(BB*NN)/256, 256, 0, stream>>>(fg_cl, unk, scales, Qm, ksArr, diagA);
    awpack_kernel<<<(BB*VD*NN)/256, 256, 0, stream>>>(alpha, awT);
    dim3 fgrid(NN/128, BB);
    flash_attn_kernel<<<fgrid, 256, 0, stream>>>(Qm, awT, Pq, ksArr, diagA, Z);
    paste_kernel<<<(BB*HF*HF)/256, 256, 0, stream>>>(Z, coefA, coefB, alpha, out);
}
